// PSMNISTSNN_15831249453628
// MI455X (gfx1250) — matrix operand order repaired and checked by reading, not compiled, not run on hardware
//
#include <hip/hip_runtime.h>

// ---------------------------------------------------------------------------
// AdLIF SNN scan for MI455X (gfx1250).
//   - Batch-independent recurrence -> 1 workgroup per 16-batch tile,
//     full T=784 time loop inside the block with workgroup barriers.
//   - Recurrent matmul spk @ Wrec.T via v_wmma_f32_16x16x32_bf16
//     (spikes are exactly representable in bf16; accum fp32).
//   - v1/a1 membrane state lives in WMMA accumulator register layout for
//     the whole scan; only the 16x512 spike tile goes through LDS.
//   - fc2 (H=512 -> O=10) is one zero-padded 16x16 WMMA tile on wave 0,
//     pipelined one step behind + epilogue.
// ---------------------------------------------------------------------------

typedef __attribute__((ext_vector_type(16))) __bf16 v16bf;
typedef __attribute__((ext_vector_type(8)))  __bf16 v8bf;
typedef __attribute__((ext_vector_type(8)))  float  v8f;

#define T_STEPS 784
#define H_DIM   512
#define O_DIM   10
#define B_DIM   512
#define MTILE   16
#define SPK_STRIDE 520   // 512 + 8 pad (ushorts); 520*2B row stride, 16B aligned

__device__ __forceinline__ unsigned short f2bf_rne(float f) {
  union { float f; unsigned int u; } x; x.f = f;
  unsigned int u = x.u;
  return (unsigned short)((u + 0x7FFFu + ((u >> 16) & 1u)) >> 16);
}

// A-matrix fragment (16x32 bf16, M x K):
//  lanes 0-15 : M = lane,      elems 0..15 = K k0+0..15 (contiguous)
//  lanes16-31 : M = lane-16,   elems 0..15 = K k0+16..31 (contiguous)
__device__ __forceinline__ v16bf load_a_frag(const unsigned short* rowp, int k0, int half) {
  const v8bf lo = *reinterpret_cast<const v8bf*>(rowp + k0 + half * 16);
  const v8bf hi = *reinterpret_cast<const v8bf*>(rowp + k0 + half * 16 + 8);
  v16bf r;
#pragma unroll
  for (int i = 0; i < 8; ++i) { r[i] = lo[i]; r[i + 8] = hi[i]; }
  return r;
}

// B-matrix fragment (32x16 bf16, K x N):
//  lane gives N = lane%16; K = k0 + 16*(lane/16) + 0..15 (contiguous).
//  colp points at row n of an [N][K] (K-contiguous) bf16 weight array.
__device__ __forceinline__ v16bf load_b_frag(const unsigned short* colp, int k0, int half) {
  const v8bf lo = *reinterpret_cast<const v8bf*>(colp + k0 + half * 16);
  const v8bf hi = *reinterpret_cast<const v8bf*>(colp + k0 + half * 16 + 8);
  v16bf r;
#pragma unroll
  for (int i = 0; i < 8; ++i) { r[i] = lo[i]; r[i + 8] = hi[i]; }
  return r;
}

// --------------------------- weight prep kernel ----------------------------
// wrecB[n][k] = bf16(Wrec[n][k])           (512 x 512)
// w2B  [n][k] = bf16(W2[n][k]) for n<10, 0 otherwise   (16 x 512, padded)
__global__ void prep_weights(const float* __restrict__ Wrec,
                             const float* __restrict__ W2,
                             unsigned short* __restrict__ wrecB,
                             unsigned short* __restrict__ w2B) {
  int idx = blockIdx.x * blockDim.x + threadIdx.x;
  if (idx < H_DIM * H_DIM) {
    wrecB[idx] = f2bf_rne(Wrec[idx]);
  }
  int i2 = idx - H_DIM * H_DIM;
  if (i2 >= 0 && i2 < 16 * H_DIM) {
    int n = i2 >> 9, k = i2 & (H_DIM - 1);
    w2B[i2] = (n < O_DIM) ? f2bf_rne(W2[n * H_DIM + k]) : (unsigned short)0;
  }
}

// ------------------------------ main kernel --------------------------------
__global__ __launch_bounds__(256) void snn_scan_kernel(
    const float* __restrict__ x,        // [B][T]
    const float* __restrict__ W1,       // [H]
    const float* __restrict__ alpha,    // [H]
    const float* __restrict__ rho,      // [H]
    const float* __restrict__ beta_a,   // [H]
    const float* __restrict__ beta_out, // [O]
    const unsigned short* __restrict__ wrecB, // [H][H] bf16 bits
    const unsigned short* __restrict__ w2B,   // [16][H] bf16 bits
    float* __restrict__ out)            // [B][O]
{
  __shared__ __align__(16) unsigned short spk[MTILE * SPK_STRIDE];
  __shared__ float xsh[MTILE];

  const int tid  = threadIdx.x;
  const int wave = tid >> 5;       // 8 waves; wave w owns N in [64w, 64w+64)
  const int lane = tid & 31;
  const int half = lane >> 4;      // lane half-group
  const int lcol = lane & 15;
  const int b0   = blockIdx.x * MTILE;

  // zero spike state, preload x(t=0)
  for (int i = tid; i < MTILE * SPK_STRIDE; i += blockDim.x) spk[i] = 0;
  if (tid < MTILE) xsh[tid] = x[(b0 + tid) * T_STEPS + 0];

  // per-n parameters for this lane's 4 N-tiles (n fixed per lane column)
  float alpv[4], omav[4], rhov[4], bav[4], w1v[4];
#pragma unroll
  for (int j = 0; j < 4; ++j) {
    const int n = wave * 64 + j * 16 + lcol;
    alpv[j] = alpha[n]; omav[j] = 1.0f - alpv[j];
    rhov[j] = rho[n];   bav[j]  = beta_a[n];  w1v[j] = W1[n];
  }
  float boutv = 0.0f;
  if (wave == 0 && lcol < O_DIM) boutv = beta_out[lcol];

  const v8f zerov = {};
  v8f v1[4] = {zerov, zerov, zerov, zerov};   // membrane, C/D register layout
  v8f a1[4] = {zerov, zerov, zerov, zerov};   // adaptation
  v8f v2 = zerov, osum = zerov;               // wave-0 readout state

  const unsigned short* arow = spk + lcol * SPK_STRIDE;  // A-frag row base

  __syncthreads();

  for (int t = 0; t < T_STEPS; ++t) {
    // ---------------- read phase: I1 = spk(t-1) @ Wrec.T -----------------
    v8f acc[4] = {zerov, zerov, zerov, zerov};
#pragma unroll 4
    for (int ki = 0; ki < H_DIM / 32; ++ki) {
      const int k0 = ki * 32;
      const v16bf af = load_a_frag(arow, k0, half);
#pragma unroll
      for (int j = 0; j < 4; ++j) {
        const int n = wave * 64 + j * 16 + lcol;
        const v16bf bf = load_b_frag(wrecB + n * H_DIM, k0, half);
        acc[j] = __builtin_amdgcn_wmma_f32_16x16x32_bf16(
            false, af, false, bf, (short)0, acc[j], false, false);
      }
    }

    // fc2 on spk(t-1) (wave 0 only; t=0 contributes exact zeros -> harmless)
    if (wave == 0) {
      v8f facc = {};
#pragma unroll 4
      for (int ki = 0; ki < H_DIM / 32; ++ki) {
        const int k0 = ki * 32;
        const v16bf af = load_a_frag(arow, k0, half);
        const v16bf bf = load_b_frag(w2B + lcol * H_DIM, k0, half);
        facc = __builtin_amdgcn_wmma_f32_16x16x32_bf16(
            false, af, false, bf, (short)0, facc, false, false);
      }
#pragma unroll
      for (int r = 0; r < 8; ++r) {
        v2[r]   = boutv * v2[r] + (1.0f - boutv) * facc[r];
        osum[r] = osum[r] + v2[r];
      }
    }

    // per-batch-row input current x[m] (m = r + 8*half in C/D layout)
    float xv[8];
#pragma unroll
    for (int r = 0; r < 8; ++r) xv[r] = xsh[r + 8 * half];

    // AdLIF pointwise update in accumulator layout
    unsigned short sbits[4][8];
#pragma unroll
    for (int j = 0; j < 4; ++j) {
#pragma unroll
      for (int r = 0; r < 8; ++r) {
        const float I1 = acc[j][r] + xv[r] * w1v[j];
        float v = alpv[j] * v1[j][r] + omav[j] * (I1 - a1[j][r]);
        const float s = (v > 1.0f) ? 1.0f : 0.0f;     // spike, threshold 1.0
        v -= s;                                       // soft reset
        v1[j][r] = v;
        a1[j][r] = rhov[j] * a1[j][r] + bav[j] * s;
        sbits[j][r] = (s > 0.0f) ? (unsigned short)0x3F80 : (unsigned short)0;
      }
    }

    __syncthreads();   // everyone done reading spk(t-1) / xsh(t)

    // ---------------- write phase: publish spk(t), stage x(t+1) ----------
#pragma unroll
    for (int j = 0; j < 4; ++j) {
      const int n = wave * 64 + j * 16 + lcol;
#pragma unroll
      for (int r = 0; r < 8; ++r)
        spk[(r + 8 * half) * SPK_STRIDE + n] = sbits[j][r];
    }
    if (tid < MTILE && t + 1 < T_STEPS)
      xsh[tid] = x[(b0 + tid) * T_STEPS + (t + 1)];
    __syncthreads();
  }

  // ---------------- epilogue: fc2 on spk(T-1), write output ---------------
  if (wave == 0) {
    v8f facc = {};
#pragma unroll 4
    for (int ki = 0; ki < H_DIM / 32; ++ki) {
      const int k0 = ki * 32;
      const v16bf af = load_a_frag(arow, k0, half);
      const v16bf bf = load_b_frag(w2B + lcol * H_DIM, k0, half);
      facc = __builtin_amdgcn_wmma_f32_16x16x32_bf16(
          false, af, false, bf, (short)0, facc, false, false);
    }
#pragma unroll
    for (int r = 0; r < 8; ++r) {
      v2[r]   = boutv * v2[r] + (1.0f - boutv) * facc[r];
      osum[r] = osum[r] + v2[r];
    }
    if (lcol < O_DIM) {
#pragma unroll
      for (int r = 0; r < 8; ++r)
        out[(b0 + r + 8 * half) * O_DIM + lcol] =
            osum[r] * (1.0f / (float)T_STEPS);
    }
  }
}

// ------------------------------- launcher ----------------------------------
extern "C" void kernel_launch(void* const* d_in, const int* in_sizes, int n_in,
                              void* d_out, int out_size, void* d_ws, size_t ws_size,
                              hipStream_t stream) {
  const float* x        = (const float*)d_in[0];
  const float* W1       = (const float*)d_in[1];
  const float* Wrec     = (const float*)d_in[2];
  const float* W2       = (const float*)d_in[3];
  const float* alpha    = (const float*)d_in[4];
  const float* rho      = (const float*)d_in[5];
  const float* beta_a   = (const float*)d_in[6];
  const float* beta_out = (const float*)d_in[7];

  unsigned short* wrecB = (unsigned short*)d_ws;              // 512*512 bf16
  unsigned short* w2B   = wrecB + H_DIM * H_DIM;              // 16*512 bf16

  const int prep_elems = H_DIM * H_DIM + 16 * H_DIM;
  prep_weights<<<(prep_elems + 255) / 256, 256, 0, stream>>>(Wrec, W2, wrecB, w2B);

  snn_scan_kernel<<<B_DIM / MTILE, 256, 0, stream>>>(
      x, W1, alpha, rho, beta_a, beta_out, wrecB, w2B, (float*)d_out);
}
